// ContinuousEnergyPredictor_17617955848231
// MI455X (gfx1250) — compile-verified, matrix operand order checked by reading, not run on hardware
//
#include <hip/hip_runtime.h>
#include <hip/hip_bf16.h>
#include <math.h>

#define B_    16
#define N_    24
#define F_    128
#define FE_   128
#define C_    27
#define M_REAL 648      // N_ * C_
#define M_PAD  656      // padded to 41 * 16
#define MT_    41
#define ZDIM_  384
#define BN_    384      // B_ * N_

// LDS A-tile row stride in halfs: 128 + 8 pad => 272B/row, bank advance 4/row
#define AROW_  136

#define LOG2E_ 1.4426950408889634f
#define LN2_   0.6931471805599453f

typedef __attribute__((ext_vector_type(16))) _Float16 v16h;
typedef __attribute__((ext_vector_type(8)))  _Float16 v8h;
typedef __attribute__((ext_vector_type(8)))  float    v8f;

// Hardware transcendentals: v_exp_f32 (exp2), v_log_f32 (log2), v_rcp_f32.
__device__ __forceinline__ float sigmoidf_(float x) {
  return __builtin_amdgcn_rcpf(1.f + __builtin_amdgcn_exp2f(x * -LOG2E_));
}
__device__ __forceinline__ float softplusf_(float x) {
  float e = __builtin_amdgcn_exp2f(-fabsf(x) * LOG2E_);
  return fmaxf(x, 0.f) + LN2_ * __builtin_amdgcn_logf(1.f + e);
}

// ---------------------------------------------------------------------------
// Kernel 1: geometry -> edge_w (f32) and edge_attr (f16, WMMA-ready rows)
// ---------------------------------------------------------------------------
__global__ void edge_kernel(const float* __restrict__ pos,
                            const float* __restrict__ cell,
                            _Float16* __restrict__ EA,
                            float* __restrict__ EW) {
  int id = blockIdx.x * blockDim.x + threadIdx.x;
  if (id >= BN_ * M_PAD) return;
  int bi = id / M_PAD;
  int m  = id - bi * M_PAD;
  int b  = bi / N_;
  int i  = bi - b * N_;
  _Float16* row = EA + (size_t)id * FE_;

  if (m >= M_REAL) {               // padding: zero attr, zero weight
    EW[id] = 0.f;
    v8h z = {};
    for (int k = 0; k < FE_; k += 8) *(v8h*)(row + k) = z;
    return;
  }
  int j = m / C_;
  int c = m - j * C_;
  float gx = (float)(c / 9) - 1.f;
  float gy = (float)((c / 3) % 3) - 1.f;
  float gz = (float)(c % 3) - 1.f;

  const float* cb = cell + b * 9;
  const float* pi = pos + (size_t)(b * N_ + i) * 3;
  const float* pj = pos + (size_t)(b * N_ + j) * 3;
  float dsqr = 0.f;
#pragma unroll
  for (int e = 0; e < 3; ++e) {
    float pbc = gx * cb[e] + gy * cb[3 + e] + gz * cb[6 + e];
    float d = pi[e] - (pj[e] + pbc);
    dsqr += d * d;
  }
  float crx = cb[4] * cb[8] - cb[5] * cb[7];
  float cry = cb[5] * cb[6] - cb[3] * cb[8];
  float crz = cb[3] * cb[7] - cb[4] * cb[6];
  float vol = cb[0] * crx + cb[1] * cry + cb[2] * crz;
  float radius = cbrtf(fabsf(vol) / (float)N_);

  bool mask = (dsqr <= radius * radius) && (dsqr > 1e-4f);
  float dist = sqrtf(fmaxf(dsqr, 1e-12f));
  float ew = mask ? (cosf(dist * 3.14159265358979323846f / radius) + 1.f) : 0.f;
  EW[id] = ew;

  const float step   = 6.0f / 127.0f;
  const float coeff2 = (-0.5f / (step * step)) * LOG2E_;  // exp(x) = exp2(x*log2e)
  for (int k0 = 0; k0 < FE_; k0 += 8) {
    v8h pack;
#pragma unroll
    for (int q = 0; q < 8; ++q) {
      float d = dist - (float)(k0 + q) * step;
      pack[q] = (_Float16)__builtin_amdgcn_exp2f(coeff2 * d * d);
    }
    *(v8h*)(row + k0) = pack;
  }
}

// ---------------------------------------------------------------------------
// Kernel 2: feat = tanh(emb[z-1])
// ---------------------------------------------------------------------------
__global__ void feat_init_kernel(const float* __restrict__ emb,
                                 const int* __restrict__ anum,
                                 float* __restrict__ feat) {
  int id = blockIdx.x * blockDim.x + threadIdx.x;
  if (id >= BN_ * F_) return;
  int row = id >> 7;
  int f   = id & 127;
  int zi  = anum[row] - 1;
  feat[id] = tanhf(emb[zi * F_ + f]);
}

// ---------------------------------------------------------------------------
// Kernel 3a: per-atom projections gi/gj/si/sj = feat @ W[:F], W[F:2F]
// ---------------------------------------------------------------------------
__global__ void proj_kernel(const float* __restrict__ feat,
                            const float* __restrict__ Wf,
                            const float* __restrict__ Ws,
                            float* __restrict__ gi, float* __restrict__ gj,
                            float* __restrict__ si, float* __restrict__ sj) {
  int id = blockIdx.x * blockDim.x + threadIdx.x;
  if (id >= BN_ * F_) return;
  int row = id >> 7;
  int f   = id & 127;
  const float* fr = feat + (size_t)row * F_;
  float a0 = 0.f, a1 = 0.f, a2 = 0.f, a3 = 0.f;
  for (int k = 0; k < F_; ++k) {
    float fv = fr[k];
    a0 += fv * Wf[k * F_ + f];
    a1 += fv * Wf[(F_ + k) * F_ + f];
    a2 += fv * Ws[k * F_ + f];
    a3 += fv * Ws[(F_ + k) * F_ + f];
  }
  gi[id] = a0; gj[id] = a1; si[id] = a2; sj[id] = a3;
}

// ---------------------------------------------------------------------------
// Kernel 3b: fused edge GEMM (WMMA f16) + gate/filter + edge reduce + feat upd
// One workgroup (8 waves) per (b,i). Wave w owns output columns [16w,16w+16).
// A-tiles staged global->LDS with async-to-LDS copies, double buffered.
// ---------------------------------------------------------------------------
__global__ void __launch_bounds__(256) conv_kernel(
    const _Float16* __restrict__ EA, const float* __restrict__ EW,
    const float* __restrict__ gi, const float* __restrict__ gj,
    const float* __restrict__ si, const float* __restrict__ sj,
    const float* __restrict__ Wf_e, const float* __restrict__ Ws_e,  // 128x128 (k,n)
    const float* __restrict__ bf, const float* __restrict__ bs,
    const float* __restrict__ featIn, float* __restrict__ featOut) {
  __shared__ float gjS[25 * F_];
  __shared__ float sjS[25 * F_];
  __shared__ float giS[F_], siS[F_], bfS[F_], bsS[F_];
  __shared__ float ewS[M_PAD];
  __shared__ int   jTab[M_PAD];
  __shared__ __align__(16) _Float16 aBuf[2][16 * AROW_];  // padded rows

  int tid = threadIdx.x;
  int bi  = blockIdx.x;
  int b   = bi / N_;

  // ---- preamble: gate tables, edge weights, j-table into LDS ----
  for (int idx = tid; idx < 25 * F_; idx += 256) {
    int j2 = idx >> 7, nn = idx & 127;
    gjS[idx] = (j2 < N_) ? gj[(b * N_ + j2) * F_ + nn] : 0.f;
    sjS[idx] = (j2 < N_) ? sj[(b * N_ + j2) * F_ + nn] : 0.f;
  }
  for (int idx = tid; idx < F_; idx += 256) {
    giS[idx] = gi[bi * F_ + idx];
    siS[idx] = si[bi * F_ + idx];
    bfS[idx] = bf[idx];
    bsS[idx] = bs[idx];
  }
  for (int idx = tid; idx < M_PAD; idx += 256) {
    ewS[idx]  = EW[bi * M_PAD + idx];
    jTab[idx] = (idx / C_) * F_;      // row offset into gjS/sjS (row 24 = zeros)
  }

  int wave = tid >> 5;
  int lane = tid & 31;
  bool hi  = lane >= 16;
  int  n   = (wave << 4) + (lane & 15);   // output column owned by this lane

  // ---- B fragments (f32 -> f16) once; K halves per ISA layout ----
  v16h bF[4], bS[4];
#pragma unroll
  for (int t = 0; t < 4; ++t) {
    int kb = t * 32 + (hi ? 8 : 0);
    v16h rf, rs;
#pragma unroll
    for (int q = 0; q < 8; ++q) {
      rf[q]     = (_Float16)Wf_e[(kb + q) * F_ + n];
      rf[8 + q] = (_Float16)Wf_e[(kb + 16 + q) * F_ + n];
      rs[q]     = (_Float16)Ws_e[(kb + q) * F_ + n];
      rs[8 + q] = (_Float16)Ws_e[(kb + 16 + q) * F_ + n];
    }
    bF[t] = rf; bS[t] = rs;
  }

  // ---- async A-tile staging setup ----
  // 256 threads x 16B = one 4KB tile per async instruction group (1 instr/wave).
  const _Float16* EAb = EA + (size_t)bi * M_PAD * FE_;
  int chunk = tid;                 // 0..255
  int crow  = chunk >> 4;          // tile row 0..15
  int cseg  = chunk & 15;          // 16B segment within row
  const _Float16* gsrcBase = EAb + (size_t)crow * FE_ + cseg * 8;
  unsigned ldst0 = (unsigned)(unsigned long long)&aBuf[0][crow * AROW_ + cseg * 8];
  unsigned ldst1 = (unsigned)(unsigned long long)&aBuf[1][crow * AROW_ + cseg * 8];

  // prefetch tile 0 into buffer 0
  {
    const _Float16* g = gsrcBase;
    asm volatile("global_load_async_to_lds_b128 %0, %1, off"
                 :: "v"(ldst0), "v"(g) : "memory");
  }
  __syncthreads();   // preamble LDS tables ready

  float gib = giS[n] + bfS[n];
  float sib = siS[n] + bsS[n];
  float msgAcc = 0.f;

  for (int mt = 0; mt < MT_; ++mt) {
    // prefetch next tile into the other buffer, then wait for current tile
    if (mt + 1 < MT_) {
      const _Float16* g = gsrcBase + (size_t)(mt + 1) * 16 * FE_;
      unsigned d = ((mt + 1) & 1) ? ldst1 : ldst0;
      asm volatile("global_load_async_to_lds_b128 %0, %1, off"
                   :: "v"(d), "v"(g) : "memory");
      asm volatile("s_wait_asynccnt 0x1" ::: "memory");  // oldest (tile mt) done
    } else {
      asm volatile("s_wait_asynccnt 0x0" ::: "memory");
    }
    __syncthreads();  // tile mt visible to all waves

    const _Float16* tile = aBuf[mt & 1];
    const _Float16* rowP = tile + (lane & 15) * AROW_;
    v8f accF = {};
    v8f accS = {};
#pragma unroll
    for (int t = 0; t < 4; ++t) {
      const _Float16* p = rowP + t * 32 + (hi ? 8 : 0);
      v8h lo8 = *(const v8h*)p;
      v8h hi8 = *(const v8h*)(p + 16);
      v16h a;
#pragma unroll
      for (int q = 0; q < 8; ++q) { a[q] = lo8[q]; a[8 + q] = hi8[q]; }
      accF = __builtin_amdgcn_wmma_f32_16x16x32_f16(false, a, false, bF[t],
                                                    (short)0, accF, false, false);
      accS = __builtin_amdgcn_wmma_f32_16x16x32_f16(false, a, false, bS[t],
                                                    (short)0, accS, false, false);
    }
    // elementwise gate/filter, weight by edge_w, reduce over the 8 rows
    int mrow = mt * 16 + (hi ? 8 : 0);
#pragma unroll
    for (int v = 0; v < 8; ++v) {
      int m = mrow + v;
      float ew  = ewS[m];
      int  joff = jTab[m];
      float g  = sigmoidf_(gib + gjS[joff + n] + accF[v]);
      float sc = softplusf_(sib + sjS[joff + n] + accS[v]);
      msgAcc = fmaf(g * sc, ew, msgAcc);
    }
    __syncthreads();  // all waves done with this buffer before it is re-filled
  }

  // combine M halves (lanes L and L+16 cover the same column n)
  float tot = msgAcc + __shfl_xor(msgAcc, 16, 32);
  if (!hi) {
    float x = featIn[bi * F_ + n] + tot;
    featOut[bi * F_ + n] = softplusf_(x);
  }
}

// ---------------------------------------------------------------------------
// Kernel 4: mean-pool + 3 spectrally-normalized FC + linear out (1 block)
// ---------------------------------------------------------------------------
__global__ void __launch_bounds__(128) head_kernel(
    const float* __restrict__ feat,
    const float* __restrict__ fcW, const float* __restrict__ fcB,
    const float* __restrict__ Wout, const float* __restrict__ bout,
    float* __restrict__ out) {
  __shared__ float hA[B_ * F_], hB[B_ * F_];
  __shared__ float uS[F_], vS[F_], red[F_];
  int t = threadIdx.x;  // 0..127

  for (int b = 0; b < B_; ++b) {
    float sm = 0.f;
    for (int nn = 0; nn < N_; ++nn) sm += feat[(b * N_ + nn) * F_ + t];
    hA[b * F_ + t] = sm * (1.f / (float)N_);
  }
  float* hin = hA;
  float* hout = hB;

  for (int l = 0; l < 3; ++l) {
    const float* W = fcW + (size_t)l * F_ * F_;
    uS[t] = 0.08838834764831845f;  // 1/sqrt(128)
    __syncthreads();
    for (int it = 0; it < 5; ++it) {
      float a = 0.f;
      for (int k = 0; k < F_; ++k) a += W[t * F_ + k] * uS[k];
      red[t] = a * a; __syncthreads();
      for (int ss = 64; ss > 0; ss >>= 1) { if (t < ss) red[t] += red[t + ss]; __syncthreads(); }
      float nv = sqrtf(red[0]) + 1e-12f;
      __syncthreads();
      vS[t] = a / nv;
      __syncthreads();
      float c = 0.f;
      for (int k = 0; k < F_; ++k) c += W[k * F_ + t] * vS[k];
      red[t] = c * c; __syncthreads();
      for (int ss = 64; ss > 0; ss >>= 1) { if (t < ss) red[t] += red[t + ss]; __syncthreads(); }
      float nu = sqrtf(red[0]) + 1e-12f;
      __syncthreads();
      uS[t] = c / nu;
      __syncthreads();
    }
    float a = 0.f;
    for (int k = 0; k < F_; ++k) a += W[t * F_ + k] * uS[k];
    red[t] = vS[t] * a; __syncthreads();
    for (int ss = 64; ss > 0; ss >>= 1) { if (t < ss) red[t] += red[t + ss]; __syncthreads(); }
    float inv = 1.f / red[0];
    __syncthreads();
    for (int b = 0; b < B_; ++b) {
      float acc = 0.f;
      for (int k = 0; k < F_; ++k) acc += hin[b * F_ + k] * W[k * F_ + t];
      hout[b * F_ + t] = softplusf_(acc * inv + fcB[l * F_ + t]);
    }
    __syncthreads();
    float* tmp = hin; hin = hout; hout = tmp;
  }
  if (t < B_) {
    float acc = 0.f;
    for (int k = 0; k < F_; ++k) acc += hin[t * F_ + k] * Wout[k];
    out[t] = acc + bout[0];
  }
}

// ---------------------------------------------------------------------------
extern "C" void kernel_launch(void* const* d_in, const int* in_sizes, int n_in,
                              void* d_out, int out_size, void* d_ws, size_t ws_size,
                              hipStream_t stream) {
  const float* pos    = (const float*)d_in[0];
  const float* cell   = (const float*)d_in[1];
  const float* emb    = (const float*)d_in[2];
  const float* convWf = (const float*)d_in[3];
  const float* convbf = (const float*)d_in[4];
  const float* convWs = (const float*)d_in[5];
  const float* convbs = (const float*)d_in[6];
  const float* fcW    = (const float*)d_in[7];
  const float* fcB    = (const float*)d_in[8];
  const float* Wout   = (const float*)d_in[9];
  const float* bout   = (const float*)d_in[10];
  const int*   anum   = (const int*)d_in[11];
  float* out = (float*)d_out;
  (void)in_sizes; (void)n_in; (void)out_size; (void)ws_size;

  char* ws = (char*)d_ws;
  size_t off = 0;
  _Float16* EA = (_Float16*)(ws + off); off += (size_t)BN_ * M_PAD * FE_ * 2;  // 64.5 MB
  float* EW    = (float*)(ws + off);    off += (size_t)BN_ * M_PAD * 4;
  float* featA = (float*)(ws + off);    off += (size_t)BN_ * F_ * 4;
  float* featB = (float*)(ws + off);    off += (size_t)BN_ * F_ * 4;
  float* gi    = (float*)(ws + off);    off += (size_t)BN_ * F_ * 4;
  float* gj    = (float*)(ws + off);    off += (size_t)BN_ * F_ * 4;
  float* si    = (float*)(ws + off);    off += (size_t)BN_ * F_ * 4;
  float* sj    = (float*)(ws + off);    off += (size_t)BN_ * F_ * 4;

  edge_kernel<<<(BN_ * M_PAD + 255) / 256, 256, 0, stream>>>(pos, cell, EA, EW);
  feat_init_kernel<<<(BN_ * F_ + 255) / 256, 256, 0, stream>>>(emb, anum, featA);

  float* fin = featA;
  float* fout = featB;
  for (int l = 0; l < 3; ++l) {
    const float* Wf = convWf + (size_t)l * ZDIM_ * F_;
    const float* Ws = convWs + (size_t)l * ZDIM_ * F_;
    proj_kernel<<<(BN_ * F_ + 255) / 256, 256, 0, stream>>>(fin, Wf, Ws, gi, gj, si, sj);
    conv_kernel<<<BN_, 256, 0, stream>>>(EA, EW, gi, gj, si, sj,
                                         Wf + 2 * F_ * F_, Ws + 2 * F_ * F_,
                                         convbf + l * F_, convbs + l * F_,
                                         fin, fout);
    float* tmp = fin; fin = fout; fout = tmp;
  }
  head_kernel<<<1, 128, 0, stream>>>(fin, fcW, fcB, Wout, bout, out);
}
